// SAGE_48301202210899
// MI455X (gfx1250) — matrix operand order repaired and checked by reading, not compiled, not run on hardware
//
#include <hip/hip_runtime.h>
#include <hip/hip_fp16.h>

typedef _Float16 v16h __attribute__((ext_vector_type(16)));
typedef _Float16 v8h  __attribute__((ext_vector_type(8)));
typedef float    v8f  __attribute__((ext_vector_type(8)));
typedef int      si4  __attribute__((ext_vector_type(4)));
typedef int      si8  __attribute__((ext_vector_type(8)));

#define HF 128  // feature width (F_IN == H == 128)

// ---------------------------------------------------------------- utilities
__global__ void zero_f32_kernel(float* __restrict__ p, int n) {
    int t = blockIdx.x * blockDim.x + threadIdx.x;
    if (t < n) p[t] = 0.0f;
}

__global__ void cvt_w_kernel(const float* __restrict__ w, _Float16* __restrict__ o, int n) {
    int t = blockIdx.x * blockDim.x + threadIdx.x;
    if (t < n) o[t] = (_Float16)w[t];
}

// h16 = (f16) log(x + 1)
__global__ void log1p_kernel(const float* __restrict__ x, _Float16* __restrict__ h16, int n) {
    int t = blockIdx.x * blockDim.x + threadIdx.x;
    if (t < n) h16[t] = (_Float16)__logf(x[t] + 1.0f);
}

// ---------------------------------------------------------------- degree
__global__ void deg_kernel(const int* __restrict__ dst, float* __restrict__ deg, int e) {
    int t = blockIdx.x * blockDim.x + threadIdx.x;
    if (t < e) atomicAdd(deg + dst[t], 1.0f);
}

__global__ void invdeg_kernel(float* __restrict__ deg, int n) {
    int t = blockIdx.x * blockDim.x + threadIdx.x;
    if (t < n) deg[t] = 1.0f / fmaxf(deg[t], 1.0f);
}

// ------------------------------------------------- edge scatter (mean-agg numerator)
// one wave (32 lanes) per edge, 4 features per lane; L2-resident f32 atomics
__global__ void scatter_kernel(const _Float16* __restrict__ h16,
                               const int* __restrict__ src, const int* __restrict__ dst,
                               float* __restrict__ msg, int e) {
    long t = (long)blockIdx.x * blockDim.x + threadIdx.x;
    long total = (long)e * 32;
    if (t >= total) return;
    int edge = (int)(t >> 5);
    int lane = (int)(t & 31);
    int s = src[edge], d = dst[edge];
    const _Float16* p = h16 + (size_t)s * HF + lane * 4;
    float* q = msg + (size_t)d * HF + lane * 4;
    atomicAdd(q + 0, (float)p[0]);
    atomicAdd(q + 1, (float)p[1]);
    atomicAdd(q + 2, (float)p[2]);
    atomicAdd(q + 3, (float)p[3]);
}

// hn16 = (f16)(msg * inv_deg)
__global__ void neigh_kernel(const float* __restrict__ msg, const float* __restrict__ invdeg,
                             _Float16* __restrict__ hn16, int n_elems) {
    int t = blockIdx.x * blockDim.x + threadIdx.x;
    if (t < n_elems) hn16[t] = (_Float16)(msg[t] * invdeg[t >> 7]);
}

// ---------------------------------------------------------------- WMMA GEMM
// r[N,128] = h16 @ Ws^T + hn16 @ Wn^T + b   (optional ReLU)
// 256 threads = 8 waves/block, 128 rows/block. Both weight matrices (contiguous
// 64 KB fp16 block) are staged into LDS once per block with a single TDM
// tensor_load_to_lds. B fragments are double-buffered from LDS; accumulators
// start from the inline-0 SRC2 form and bias is added in the store epilogue,
// keeping the WMMA stream free of broadcast/wait bubbles.
__global__ __launch_bounds__(256) void sage_gemm_kernel(
        const _Float16* __restrict__ h16, const _Float16* __restrict__ hn16,
        const _Float16* __restrict__ Wpair,   // [2][128][128] f16: self then neigh
        const float* __restrict__ bias, float* __restrict__ r,
        int n_rows, int do_relu) {
    __shared__ __align__(32) _Float16 lw[2 * HF * HF];   // 64 KB

    // ---- TDM stage: one descriptor copies both matrices (32768 f16) ----
    if (threadIdx.x < 32) {                      // wave 0 only (TDM ignores EXEC)
        unsigned lds_base = (unsigned)(size_t)(void*)lw;           // low 32 = LDS offset
        unsigned long long ga = (unsigned long long)(size_t)Wpair; // 57-bit global addr
        si4 g0;
        g0.x = 1;                                              // count=1, user D#
        g0.y = (int)lds_base;                                  // lds_addr
        g0.z = (int)(unsigned)(ga & 0xFFFFFFFFu);              // global_addr[31:0]
        g0.w = (int)(((unsigned)(ga >> 32) & 0x01FFFFFFu)      // global_addr[56:32]
                     | 0x80000000u);                           // type=2 (image)
        si8 g1;
        g1[0] = 0x00010000;            // workgroup_mask=0, data_size=1 (2 bytes)
        g1[1] = (int)(32768u << 16);   // tensor_dim0 = 32768 (low16 at [63:48])
        g1[2] = 1 << 16;               // tensor_dim0 hi = 0, tensor_dim1 = 1
        g1[3] = (int)(32768u << 16);   // tensor_dim1 hi = 0, tile_dim0 = 32768
        g1[4] = 1;                     // tile_dim1 = 1, tile_dim2 = 0
        g1[5] = 32768;                 // tensor_dim0_stride low32
        g1[6] = 0;
        g1[7] = 0;
        asm volatile("tensor_load_to_lds %0, %1" :: "s"(g0), "s"(g1) : "memory");
        __builtin_amdgcn_s_wait_tensorcnt(0);
    }
    __syncthreads();

    const int lane = threadIdx.x & 31;
    const int wave = threadIdx.x >> 5;
    int m_base = (blockIdx.x * 8 + wave) * 16;
    if (m_base + 16 > n_rows) m_base = n_rows - 16;   // idempotent tail, EXEC stays full

    const int m  = lane & 15;
    const int kh = lane >> 4;               // which 16-lane half
    const size_t row = (size_t)(m_base + m) * HF;
    const int c0 = kh ? 16 : 0;             // contiguous K layout (k = 16*kh + s), matches B
    const int c1 = kh ? 24 : 8;

    // hoist bias loads out of the WMMA stream (one value per o-tile, per lane)
    float bv[8];
#pragma unroll
    for (int ot = 0; ot < 8; ++ot) bv[ot] = bias[ot * 16 + m];

    // Preload A fragments for both operand streams, all of K=128
    v16h as[4], an[4];
#pragma unroll
    for (int kt = 0; kt < 4; ++kt) {
        const _Float16* ph = h16  + row + kt * 32;
        const _Float16* pn = hn16 + row + kt * 32;
        v8h s0 = *(const v8h*)(ph + c0);
        v8h s1 = *(const v8h*)(ph + c1);
        v8h n0 = *(const v8h*)(pn + c0);
        v8h n1 = *(const v8h*)(pn + c1);
#pragma unroll
        for (int i = 0; i < 8; ++i) {
            as[kt][i] = s0[i]; as[kt][i + 8] = s1[i];
            an[kt][i] = n0[i]; an[kt][i + 8] = n1[i];
        }
    }

    // B-fragment fetch from LDS: mat 0 = W_self, mat 1 = W_neigh
    auto ldB = [&](int mat, int ot, int kt) -> v16h {
        return *(const v16h*)(lw + (size_t)mat * HF * HF +
                              (size_t)(ot * 16 + m) * HF + kt * 32 + kh * 16);
    };

    // Software-pipelined: prefetch next B fragments while WMMAs consume current
    v16h bs = ldB(0, 0, 0);
    v16h bn = ldB(1, 0, 0);
#pragma unroll
    for (int ot = 0; ot < 8; ++ot) {
        v8f acc = {};                       // first WMMA uses inline SRC2 = 0

#pragma unroll
        for (int kt = 0; kt < 4; ++kt) {
            const bool last = (ot == 7) && (kt == 3);
            const int no = ((kt == 3) ? ot + 1 : ot) & 7;
            const int nk = (kt == 3) ? 0 : kt + 1;
            v16h bs2 = bs, bn2 = bn;
            if (!last) { bs2 = ldB(0, no, nk); bn2 = ldB(1, no, nk); }
            acc = __builtin_amdgcn_wmma_f32_16x16x32_f16(false, as[kt], false, bs,
                                                         (short)0, acc, false, false);
            acc = __builtin_amdgcn_wmma_f32_16x16x32_f16(false, an[kt], false, bn,
                                                         (short)0, acc, false, false);
            bs = bs2; bn = bn2;
        }

        // epilogue: bias (+ReLU) off the WMMA dependency chain
#pragma unroll
        for (int j = 0; j < 8; ++j) {
            const int mm = m_base + j + kh * 8;
            float v = acc[j] + bv[ot];
            if (do_relu) v = fmaxf(v, 0.0f);
            r[(size_t)mm * HF + ot * 16 + m] = v;
        }
    }
}

// ---------------------------------------------------------------- PairNorm
// single pass: per-column sum and sum of squares
__global__ void pn_reduce_kernel(const float* __restrict__ r,
                                 float* __restrict__ colsum, float* __restrict__ colsumsq,
                                 int n_rows) {
    const int k = threadIdx.x;              // 128 threads = 128 columns
    int r0 = blockIdx.x * 64;
    int r1 = min(r0 + 64, n_rows);
    float s = 0.0f, sq = 0.0f;
    for (int mrow = r0; mrow < r1; ++mrow) {
        float v = r[(size_t)mrow * HF + k];
        s += v; sq += v * v;
    }
    atomicAdd(colsum + k, s);
    atomicAdd(colsumsq + k, sq);
}

// mean[k] = sum/N ; stats[0] = 1/sqrt( (sum_k (sumsq - N*mean^2)) / N + eps )
__global__ void pn_finalize_kernel(const float* __restrict__ colsum,
                                   const float* __restrict__ colsumsq,
                                   float* __restrict__ mean, float* __restrict__ stats,
                                   int n_rows) {
    __shared__ float sh[HF];
    const int k = threadIdx.x;
    const float s = colsum[k];
    const float mu = s / (float)n_rows;
    mean[k] = mu;
    sh[k] = colsumsq[k] - s * mu;           // = sumsq_k - N*mu_k^2
    __syncthreads();
    for (int off = 64; off > 0; off >>= 1) {
        if (k < off) sh[k] += sh[k + off];
        __syncthreads();
    }
    if (k == 0) stats[0] = rsqrtf(sh[0] / (float)n_rows + 1e-6f);
}

// next-layer features (f16): scale * (r - mean) * inv_rms
__global__ void pn_apply_kernel(const float* __restrict__ r, const float* __restrict__ mean,
                                const float* __restrict__ stats, _Float16* __restrict__ h16,
                                int n_elems) {
    int t = blockIdx.x * blockDim.x + threadIdx.x;
    if (t < n_elems) h16[t] = (_Float16)((r[t] - mean[t & 127]) * stats[0]);
}

// ---------------------------------------------------------------- launch
extern "C" void kernel_launch(void* const* d_in, const int* in_sizes, int n_in,
                              void* d_out, int out_size, void* d_ws, size_t ws_size,
                              hipStream_t stream) {
    const float* x   = (const float*)d_in[0];
    const int*   src = (const int*)d_in[1];
    const int*   dst = (const int*)d_in[2];
    const float* Wsf[3] = {(const float*)d_in[3], (const float*)d_in[6], (const float*)d_in[9]};
    const float* Wnf[3] = {(const float*)d_in[4], (const float*)d_in[7], (const float*)d_in[10]};
    const float* bf[3]  = {(const float*)d_in[5], (const float*)d_in[8], (const float*)d_in[11]};

    const int N = in_sizes[0] / HF;
    const int E = in_sizes[1];
    const int NE = N * HF;                  // 12.8M elements

    // workspace carve-out (256B aligned blocks)
    char* ws = (char*)d_ws;
    size_t off = 0;
    auto carve = [&](size_t bytes) -> void* {
        void* p = ws + off;
        off = (off + bytes + 255) & ~(size_t)255;
        return p;
    };
    _Float16* h16   = (_Float16*)carve((size_t)NE * 2);
    _Float16* hn16  = (_Float16*)carve((size_t)NE * 2);
    float*    f32b  = (float*)carve((size_t)NE * 4);    // msg, then r (aliased)
    _Float16* wh    = (_Float16*)carve((size_t)6 * HF * HF * 2);
    float*    deg   = (float*)carve((size_t)N * 4);     // becomes inv_deg in place
    float*    mean  = (float*)carve(HF * 4);
    float*    csum  = (float*)carve(HF * 4);
    float*    csq   = (float*)carve(HF * 4);
    float*    stats = (float*)carve(256);

    const int T = 256;
    const int gNE = (NE + T - 1) / T;
    const int gE  = (E + T - 1) / T;
    const int gN  = (N + T - 1) / T;
    const int gE32 = (int)(((long)E * 32 + T - 1) / T);
    const int gGemm = (N + 127) / 128;      // 128 rows per block (8 waves x 16)
    const int gRed  = (N + 63) / 64;
    const int gW = (HF * HF + T - 1) / T;

    // fp16 weight copies (self/neigh pairs contiguous per layer for one-shot TDM)
    for (int l = 0; l < 3; ++l) {
        cvt_w_kernel<<<gW, T, 0, stream>>>(Wsf[l], wh + (size_t)(2 * l + 0) * HF * HF, HF * HF);
        cvt_w_kernel<<<gW, T, 0, stream>>>(Wnf[l], wh + (size_t)(2 * l + 1) * HF * HF, HF * HF);
    }

    // h = log(x+1) (fp16 working copy)
    log1p_kernel<<<gNE, T, 0, stream>>>(x, h16, NE);

    // degree -> inverse degree (constant across layers)
    zero_f32_kernel<<<gN, T, 0, stream>>>(deg, N);
    deg_kernel<<<gE, T, 0, stream>>>(dst, deg, E);
    invdeg_kernel<<<gN, T, 0, stream>>>(deg, N);

    for (int l = 0; l < 3; ++l) {
        const _Float16* wpair = wh + (size_t)(2 * l) * HF * HF;

        // mean aggregation
        zero_f32_kernel<<<gNE, T, 0, stream>>>(f32b, NE);
        scatter_kernel<<<gE32, T, 0, stream>>>(h16, src, dst, f32b, E);
        neigh_kernel<<<gNE, T, 0, stream>>>(f32b, deg, hn16, NE);

        // fused dual-GEMM + bias (+ReLU); f32b (msg) is dead, reuse as r
        float* rdst = (l == 2) ? (float*)d_out : f32b;
        sage_gemm_kernel<<<gGemm, 256, 0, stream>>>(h16, hn16, wpair, bf[l],
                                                    rdst, N, l < 2 ? 1 : 0);

        if (l < 2) {
            // PairNorm
            zero_f32_kernel<<<1, HF, 0, stream>>>(csum, HF);
            zero_f32_kernel<<<1, HF, 0, stream>>>(csq, HF);
            pn_reduce_kernel<<<gRed, HF, 0, stream>>>(f32b, csum, csq, N);
            pn_finalize_kernel<<<1, HF, 0, stream>>>(csum, csq, mean, stats, N);
            pn_apply_kernel<<<gNE, T, 0, stream>>>(f32b, mean, stats, h16, NE);
        }
    }
}
